// BlockwiseEarlyExitMamba_26319559590236
// MI455X (gfx1250) — matrix operand order repaired and checked by reading, not compiled, not run on hardware
//
#include <hip/hip_runtime.h>
#include <hip/hip_bf16.h>

// ---------------------------------------------------------------------------
// BlockwiseEarlyExitMamba for MI455X (gfx1250, wave32, WMMA)
//
//   cvt weights->padded f16, build_cat(zero-padded K=160)
//   -> WMMA GEMM(fusion)+LN -> 4x [WMMA in_proj, conv+silu, WMMA x_proj,
//   WMMA dt_proj(+softplus), scan(+gate), WMMA out_proj, residual LN]
//   -> 3 early-exit classifiers.
//
// GEMM: f16 inputs / f32 accumulate (v_wmma_f32_16x16x32_f16, K=32/instr).
// All K dims padded to multiples of 32 (zero-padded weights/activations),
// B rows clamped + store-guarded for ragged N -> hot loop is branch-free.
// ---------------------------------------------------------------------------

typedef __attribute__((ext_vector_type(16))) _Float16 v16h;
typedef __attribute__((ext_vector_type(8)))  _Float16 v8h;
typedef __attribute__((ext_vector_type(4)))  _Float16 v4h;
typedef __attribute__((ext_vector_type(8)))  float    v8f;
typedef __attribute__((ext_vector_type(4)))  float    v4f;

#define B_   256
#define L_   64
#define DM_  256
#define DI_  512
#define DS_  16
#define DC_  4
#define DTR_ 16
#define NL_  4
#define T_   (B_ * L_)   // 16384 tokens
#define KCAT 160         // 136 padded to 32-multiple

#define ACT_NONE     0
#define ACT_SOFTPLUS 1

// ---------------------------------------------------------------------------
// Weight converter: f32 [N,K] -> f16 [N,Kpad], zero padding for k >= K.
// ---------------------------------------------------------------------------
__global__ __launch_bounds__(256)
void eemamba_cvt_w(const float* __restrict__ src, _Float16* __restrict__ dst,
                   int K, int Kpad, int total) {
  int gid = blockIdx.x * 256 + threadIdx.x;
  if (gid >= total) return;
  int row = gid / Kpad;
  int col = gid - row * Kpad;
  dst[gid] = (col < K) ? (_Float16)src[(size_t)row * K + col] : (_Float16)0.f;
}

// ---------------------------------------------------------------------------
// Kernel 0: gather embeddings + tiny projections into cat[T,KCAT] (zero pad)
// ---------------------------------------------------------------------------
__global__ void eemamba_build_cat(const float* __restrict__ x,
                                  const float* __restrict__ emb_proto,
                                  const float* __restrict__ emb_flags,
                                  const float* __restrict__ emb_dir,
                                  const float* __restrict__ plW,
                                  const float* __restrict__ plb,
                                  const float* __restrict__ piW,
                                  const float* __restrict__ pib,
                                  float* __restrict__ cat) {
  int tok = blockIdx.x;
  int j   = threadIdx.x;          // 0..159
  const float* xv = x + (size_t)tok * 5;
  float out;
  if (j < 32) {
    int proto = min(max((int)xv[0], 0), 255);
    out = emb_proto[proto * 32 + j];
  } else if (j < 64) {
    out = xv[1] * plW[j - 32] + plb[j - 32];
  } else if (j < 96) {
    int flags = min(max((int)xv[2], 0), 63);
    out = emb_flags[flags * 32 + (j - 64)];
  } else if (j < 128) {
    out = xv[3] * piW[j - 96] + pib[j - 96];
  } else if (j < 136) {
    int dir = min(max((int)xv[4], 0), 1);
    out = emb_dir[dir * 8 + (j - 128)];
  } else {
    out = 0.f;                    // K padding
  }
  cat[(size_t)tok * KCAT + j] = out;
}

// ---------------------------------------------------------------------------
// Generic WMMA GEMM: Y[T,N] = act( X[T,K] @ W16[N,K]^T + bias )
//   - K multiple of 32; X rows valid for cols [0,K)
//   - W16: padded f16 weights; B rows clamped to N-1, stores guarded on N
// 256 threads = 8 waves; block tile 32(M) x 128(N); each wave owns two 16x16
// tiles sharing one A fragment -> 2 v_wmma per K-step. Branch-free hot loop.
// ---------------------------------------------------------------------------
__global__ __launch_bounds__(256)
void eemamba_gemm_wmma(const float* __restrict__ X, int ldx, int K,
                       const _Float16* __restrict__ W16, // [N,K] f16 padded
                       const float* __restrict__ bias,   // may be null
                       float* __restrict__ Y, int N, int act) {
  __shared__ __align__(32) _Float16 As[32][32];
  __shared__ __align__(32) _Float16 Bs[128][32];

  const int tid  = threadIdx.x;
  const int lane = tid & 31;
  const int wave = tid >> 5;       // 0..7
  const int wr   = wave & 1;       // M sub-tile 0..1
  const int wc   = wave >> 1;      // N sub-tile 0..3 (second tile at +64 cols)
  const int blockM = blockIdx.y * 32;
  const int blockN = blockIdx.x * 128;

  v8f acc0 = {}, acc1 = {};
  const int Ksteps = K >> 5;

  // A staging: 4 contiguous cols per thread (32x32 tile)
  const int arow = tid >> 3;             // 0..31
  const int acol = (tid & 7) << 2;       // 0,4,..,28
  const float* aptr = X + (size_t)(blockM + arow) * ldx;

  // B staging: 16 contiguous f16 cols per thread (128x32 tile)
  const int brow = tid >> 1;             // 0..127
  const int bcol = (tid & 1) << 4;       // 0 or 16
  const int bn   = blockN + brow;
  const _Float16* bptr = W16 + (size_t)((bn < N) ? bn : (N - 1)) * K;

  for (int ks = 0; ks < Ksteps; ++ks) {
    const int kbase = ks << 5;

    // ---- stage A: 16B f32 load, packed convert, 8B LDS store ----
    v4f fa = *(const v4f*)(aptr + kbase + acol);
    v4h ha;
    ha[0] = (_Float16)fa[0]; ha[1] = (_Float16)fa[1];
    ha[2] = (_Float16)fa[2]; ha[3] = (_Float16)fa[3];
    *(v4h*)&As[arow][acol] = ha;

    // ---- stage B: 32B f16 load -> 32B LDS store (no conversion) ----
    *(v16h*)&Bs[brow][bcol] = *(const v16h*)(bptr + kbase + bcol);

    if (ks + 1 < Ksteps) {                          // uniform
      __builtin_prefetch(aptr + kbase + 32 + acol, 0, 1);
      __builtin_prefetch(bptr + kbase + 32 + bcol, 0, 1);
    }
    __syncthreads();

    // ---- fragments (contiguous k = 16h + s, same order in A and B) ----
    {
      int m   = (wr << 4) + (lane & 15);
      int ka0 = (lane >> 4) << 4;                   // 0 or 16
      v16h a  = *(const v16h*)&As[m][ka0];

      int nn  = (wc << 4) + (lane & 15);
      int kb0 = (lane >> 4) << 4;                   // 0 or 16
      v16h b0 = *(const v16h*)&Bs[nn][kb0];
      v16h b1 = *(const v16h*)&Bs[64 + nn][kb0];

      acc0 = __builtin_amdgcn_wmma_f32_16x16x32_f16(
               false, a, false, b0, (short)0, acc0, false, false);
      acc1 = __builtin_amdgcn_wmma_f32_16x16x32_f16(
               false, a, false, b1, (short)0, acc1, false, false);
    }
    __syncthreads();
  }

  // ---- store: VGPR v -> row M = v + 8*(l>=16), col N = l&15 ----
  const int mbase = blockM + (wr << 4) + ((lane >> 4) << 3);
  const int ncol0 = blockN + (wc << 4) + (lane & 15);
#pragma unroll
  for (int half = 0; half < 2; ++half) {
    int ncol = ncol0 + half * 64;
    if (ncol < N) {
      float bv = bias ? bias[ncol] : 0.f;
#pragma unroll
      for (int v = 0; v < 8; ++v) {
        float val = (half ? acc1[v] : acc0[v]) + bv;
        if (act == ACT_SOFTPLUS)
          val = (val > 20.f) ? val : log1pf(__expf(val));
        Y[(size_t)(mbase + v) * N + ncol] = val;
      }
    }
  }
}

// ---------------------------------------------------------------------------
// LayerNorm over DM=256, optional residual.  One block (256 thr) per token.
// ---------------------------------------------------------------------------
__global__ __launch_bounds__(256)
void eemamba_ln(const float* __restrict__ src, const float* resid,
                const float* __restrict__ g, const float* __restrict__ bta,
                float* dst) {
  __shared__ float sA[8], sB[8];
  int tok = blockIdx.x;
  int i   = threadIdx.x;
  float v = src[(size_t)tok * DM_ + i];
  if (resid) v += resid[(size_t)tok * DM_ + i];
  float s = v, q = v * v;
#pragma unroll
  for (int o = 16; o > 0; o >>= 1) {
    s += __shfl_xor(s, o, 32);
    q += __shfl_xor(q, o, 32);
  }
  int lane = i & 31, w = i >> 5;
  if (lane == 0) { sA[w] = s; sB[w] = q; }
  __syncthreads();
  if (i == 0) {
    float ts = 0.f, tq = 0.f;
#pragma unroll
    for (int k = 0; k < 8; ++k) { ts += sA[k]; tq += sB[k]; }
    sA[0] = ts; sB[0] = tq;
  }
  __syncthreads();
  float mean = sA[0] * (1.f / DM_);
  float var  = sB[0] * (1.f / DM_) - mean * mean;
  float inv  = rsqrtf(var + 1e-5f);
  dst[(size_t)tok * DM_ + i] = (v - mean) * inv * g[i] + bta[i];
}

// ---------------------------------------------------------------------------
// Depthwise causal conv (DC=4) + SiLU. u = xz[..., :DI], row stride 2*DI.
// ---------------------------------------------------------------------------
__global__ __launch_bounds__(256)
void eemamba_conv_silu(const float* __restrict__ xz,
                       const float* __restrict__ Wc,   // [DI,DC]
                       const float* __restrict__ bc,   // [DI]
                       float* __restrict__ uo) {       // [T,DI]
  int gid = blockIdx.x * blockDim.x + threadIdx.x;
  if (gid >= T_ * DI_) return;
  int d   = gid & (DI_ - 1);
  int tok = gid / DI_;
  int t   = tok & (L_ - 1);
  float acc = bc[d];
#pragma unroll
  for (int k = 0; k < DC_; ++k) {
    int dtt = k - (DC_ - 1);          // -3..0
    float v = xz[(size_t)(tok + ((t + dtt >= 0) ? dtt : 0)) * (2 * DI_) + d];
    acc += (t + dtt >= 0) ? v * Wc[d * DC_ + k] : 0.f;
  }
  uo[(size_t)tok * DI_ + d] = acc * (1.f / (1.f + __expf(-acc)));
}

// ---------------------------------------------------------------------------
// Selective scan: 1 block per batch element, 512 threads (one per channel d).
// h[16] kept in registers; B_t/C_t staged in LDS each step; SiLU(z) gate.
// yg may alias dt (same element read-then-written by the same thread).
// ---------------------------------------------------------------------------
__global__ __launch_bounds__(512)
void eemamba_scan(const float* __restrict__ u,     // [T,DI]
                  const float* dtp,                // [T,DI]
                  const float* __restrict__ xdbl,  // [T,48]
                  const float* __restrict__ xz,    // [T,2*DI] (z at +DI)
                  const float* __restrict__ A_log, // [DI,DS]
                  const float* __restrict__ Dp,    // [DI]
                  float* yg) {                     // [T,DI]
  __shared__ float BC[2 * DS_];
  int b = blockIdx.x;
  int d = threadIdx.x;
  float A[DS_];
#pragma unroll
  for (int n = 0; n < DS_; ++n) A[n] = -__expf(A_log[d * DS_ + n]);
  float Dd = Dp[d];
  float h[DS_];
#pragma unroll
  for (int n = 0; n < DS_; ++n) h[n] = 0.f;

  for (int t = 0; t < L_; ++t) {
    int tok = b * L_ + t;
    if (d < 2 * DS_) BC[d] = xdbl[(size_t)tok * 48 + DTR_ + d];
    __syncthreads();
    float dtv = dtp[(size_t)tok * DI_ + d];
    float uv  = u[(size_t)tok * DI_ + d];
    float du  = dtv * uv;
    float y = 0.f;
#pragma unroll
    for (int n = 0; n < DS_; ++n) {
      h[n] = __expf(dtv * A[n]) * h[n] + du * BC[n];
      y += h[n] * BC[DS_ + n];
    }
    y += uv * Dd;
    float z  = xz[(size_t)tok * (2 * DI_) + DI_ + d];
    float sz = z * (1.f / (1.f + __expf(-z)));
    float out = y * sz;
    __syncthreads();                 // BC fully consumed before next stage
    yg[(size_t)tok * DI_ + d] = out;
  }
}

// ---------------------------------------------------------------------------
// Early-exit classifiers: grid (B,3), 128 threads.
// ---------------------------------------------------------------------------
__global__ __launch_bounds__(128)
void eemamba_classifier(const float* __restrict__ feat,
                        const float* __restrict__ W1, const float* __restrict__ b1,
                        const float* __restrict__ W2, const float* __restrict__ b2,
                        float* __restrict__ out) {
  __shared__ float hbuf[DM_];
  __shared__ float hh[128];
  int b = blockIdx.x;
  int i = blockIdx.y;                       // head 0..2
  int j = threadIdx.x;                      // 0..127
  int pos = (i == 0) ? 7 : ((i == 1) ? 15 : 31);   // min(EXIT_POS, L)-1
  int tok = b * L_ + pos;
  hbuf[j]       = feat[(size_t)tok * DM_ + j];
  hbuf[j + 128] = feat[(size_t)tok * DM_ + j + 128];
  __syncthreads();
  const float* w = W1 + ((size_t)i * 128 + j) * DM_;
  float acc = b1[i * 128 + j];
  for (int k = 0; k < DM_; ++k) acc += hbuf[k] * w[k];
  hh[j] = fmaxf(acc, 0.f);
  __syncthreads();
  if (j < 2) {
    const float* w2 = W2 + ((size_t)i * 2 + j) * 128;
    float acc2 = b2[i * 2 + j];
    for (int k = 0; k < 128; ++k) acc2 += hh[k] * w2[k];
    out[((size_t)i * B_ + b) * 2 + j] = acc2;
  }
}

// ---------------------------------------------------------------------------
extern "C" void kernel_launch(void* const* d_in, const int* in_sizes, int n_in,
                              void* d_out, int out_size, void* d_ws, size_t ws_size,
                              hipStream_t stream) {
  (void)in_sizes; (void)n_in; (void)out_size; (void)ws_size;
  const float* x          = (const float*)d_in[0];
  const float* emb_proto  = (const float*)d_in[1];
  const float* emb_flags  = (const float*)d_in[2];
  const float* emb_dir    = (const float*)d_in[3];
  const float* proj_len_W = (const float*)d_in[4];
  const float* proj_len_b = (const float*)d_in[5];
  const float* proj_iat_W = (const float*)d_in[6];
  const float* proj_iat_b = (const float*)d_in[7];
  const float* fusion_W   = (const float*)d_in[8];
  const float* fusion_b   = (const float*)d_in[9];
  const float* tok_ln_g   = (const float*)d_in[10];
  const float* tok_ln_b   = (const float*)d_in[11];
  const float* in_proj_W  = (const float*)d_in[12];
  const float* conv_W     = (const float*)d_in[13];
  const float* conv_b     = (const float*)d_in[14];
  const float* x_proj_W   = (const float*)d_in[15];
  const float* dt_proj_W  = (const float*)d_in[16];
  const float* dt_proj_b  = (const float*)d_in[17];
  const float* A_log      = (const float*)d_in[18];
  const float* Dvec       = (const float*)d_in[19];
  const float* out_proj_W = (const float*)d_in[20];
  const float* norm_g     = (const float*)d_in[21];
  const float* norm_b     = (const float*)d_in[22];
  const float* cls_W1     = (const float*)d_in[23];
  const float* cls_b1     = (const float*)d_in[24];
  const float* cls_W2     = (const float*)d_in[25];
  const float* cls_b2     = (const float*)d_in[26];

  // ---- workspace layout: f32 activations, then padded f16 weights ----
  float* ws   = (float*)d_ws;
  float* cat  = ws;                               // T*KCAT
  float* feat = cat  + (size_t)T_ * KCAT;         // T*DM
  float* xz   = feat + (size_t)T_ * DM_;          // T*2DI
  float* ucv  = xz   + (size_t)T_ * 2 * DI_;      // T*DI
  float* xdbl = ucv  + (size_t)T_ * DI_;          // T*48
  float* dtb  = xdbl + (size_t)T_ * 48;           // T*DI (also yg, aliased)
  float* mout = dtb  + (size_t)T_ * DI_;          // T*DM
  _Float16* h16    = (_Float16*)(mout + (size_t)T_ * DM_);  // 64B aligned
  _Float16* fus16  = h16;                                   // 256 x 160
  _Float16* inp16  = fus16 + (size_t)DM_ * KCAT;            // 4096 x 256
  _Float16* xp16   = inp16 + (size_t)NL_ * 2 * DI_ * DM_;   // 192 x 512
  _Float16* dtp16  = xp16  + (size_t)NL_ * 48 * DI_;        // 2048 x 32
  _Float16* outp16 = dtp16 + (size_t)NL_ * DI_ * 32;        // 1024 x 512

  const dim3 blk256(256), blk512(512), blk128(128), blk160(160);
  const int mTiles = T_ / 32;                     // 512

  // ---- one-shot weight conversions (f32 -> padded f16) ----
  {
    int tot;
    tot = DM_ * KCAT;                 // fusion 256x136 -> 256x160
    eemamba_cvt_w<<<(tot + 255) / 256, blk256, 0, stream>>>(fusion_W, fus16, 136, KCAT, tot);
    tot = NL_ * 2 * DI_ * DM_;        // in_proj (4*1024)x256
    eemamba_cvt_w<<<(tot + 255) / 256, blk256, 0, stream>>>(in_proj_W, inp16, DM_, DM_, tot);
    tot = NL_ * 48 * DI_;             // x_proj (4*48)x512
    eemamba_cvt_w<<<(tot + 255) / 256, blk256, 0, stream>>>(x_proj_W, xp16, DI_, DI_, tot);
    tot = NL_ * DI_ * 32;             // dt_proj (4*512)x16 -> x32 zero-padded
    eemamba_cvt_w<<<(tot + 255) / 256, blk256, 0, stream>>>(dt_proj_W, dtp16, DTR_, 32, tot);
    tot = NL_ * DM_ * DI_;            // out_proj (4*256)x512
    eemamba_cvt_w<<<(tot + 255) / 256, blk256, 0, stream>>>(out_proj_W, outp16, DI_, DI_, tot);
  }

  // --- embedding + fusion + token LN ---
  eemamba_build_cat<<<T_, blk160, 0, stream>>>(x, emb_proto, emb_flags, emb_dir,
      proj_len_W, proj_len_b, proj_iat_W, proj_iat_b, cat);
  eemamba_gemm_wmma<<<dim3(DM_ / 128, mTiles), blk256, 0, stream>>>(
      cat, KCAT, KCAT, fus16, fusion_b, feat, DM_, ACT_NONE);
  eemamba_ln<<<T_, blk256, 0, stream>>>(feat, nullptr, tok_ln_g, tok_ln_b, feat);

  // --- 4 Mamba layers ---
  for (int l = 0; l < NL_; ++l) {
    const _Float16* Wi  = inp16  + (size_t)l * 2 * DI_ * DM_;
    const float*    Wc  = conv_W + (size_t)l * DI_ * DC_;
    const float*    bc  = conv_b + (size_t)l * DI_;
    const _Float16* Wx  = xp16   + (size_t)l * 48 * DI_;
    const _Float16* Wdt = dtp16  + (size_t)l * DI_ * 32;
    const float*    bdt = dt_proj_b + (size_t)l * DI_;
    const float*    Al  = A_log  + (size_t)l * DI_ * DS_;
    const float*    Dl  = Dvec   + (size_t)l * DI_;
    const _Float16* Wo  = outp16 + (size_t)l * DM_ * DI_;

    // xz = feat @ Wi^T                       [T, 1024]
    eemamba_gemm_wmma<<<dim3((2 * DI_) / 128, mTiles), blk256, 0, stream>>>(
        feat, DM_, DM_, Wi, nullptr, xz, 2 * DI_, ACT_NONE);
    // u = silu(causal_conv(xz[:, :DI]))      [T, 512]
    eemamba_conv_silu<<<(T_ * DI_) / 256, blk256, 0, stream>>>(xz, Wc, bc, ucv);
    // x_dbl = u @ Wx^T                       [T, 48]  (N ragged, store-guarded)
    eemamba_gemm_wmma<<<dim3(1, mTiles), blk256, 0, stream>>>(
        ucv, DI_, DI_, Wx, nullptr, xdbl, DTR_ + 2 * DS_, ACT_NONE);
    // dt = softplus(x_dbl[:, :16] @ Wdt^T + bdt)  [T, 512]
    // K=32: A cols 16..31 are B-values but Wdt cols 16..31 are zero.
    eemamba_gemm_wmma<<<dim3(DI_ / 128, mTiles), blk256, 0, stream>>>(
        xdbl, DTR_ + 2 * DS_, 32, Wdt, bdt, dtb, DI_, ACT_SOFTPLUS);
    // selective scan + D skip + silu(z) gate -> yg (aliases dtb)
    eemamba_scan<<<B_, blk512, 0, stream>>>(ucv, dtb, xdbl, xz, Al, Dl, dtb);
    // mout = yg @ Wo^T                       [T, 256]
    eemamba_gemm_wmma<<<dim3(DM_ / 128, mTiles), blk256, 0, stream>>>(
        dtb, DI_, DI_, Wo, nullptr, mout, DM_, ACT_NONE);
    // feat = LN(mout + feat)
    eemamba_ln<<<T_, blk256, 0, stream>>>(mout, feat, norm_g, norm_b, feat);
  }

  // --- early-exit heads ---
  eemamba_classifier<<<dim3(B_, 3), blk128, 0, stream>>>(
      feat, cls_W1, cls_b1, cls_W2, cls_b2, (float*)d_out);
}
